// GRUClassifier_49177375539207
// MI455X (gfx1250) — matrix operand order repaired and checked by reading, not compiled, not run on hardware
//
#include <hip/hip_runtime.h>
#include <hip/hip_bf16.h>
#include <stdint.h>

// Problem dims (fixed by the reference)
#define T_SEQ   512
#define B_BATCH 128
#define D_IN    256
#define U_HID   256
#define NG3     768           // 3*U
#define SCAN_BLOCKS 16        // 16 workgroups x 16 hidden columns each

typedef __attribute__((ext_vector_type(16))) __bf16        bf16x16;
typedef __attribute__((ext_vector_type(8)))  float         f32x8;
typedef __attribute__((ext_vector_type(4)))  unsigned int  u32x4;

union FragBF { bf16x16 v; u32x4 q[2]; };

__device__ __forceinline__ unsigned short f2bf(float f) {
  unsigned int u = __float_as_uint(f);
  u += 0x7FFFu + ((u >> 16) & 1u);          // round-to-nearest-even
  return (unsigned short)(u >> 16);
}
__device__ __forceinline__ float sigmoidf_(float x) { return 1.0f / (1.0f + __expf(-x)); }
__device__ __forceinline__ float tanhf_(float x)    { return 1.0f - 2.0f / (1.0f + __expf(2.0f * x)); }

// ---------------------------------------------------------------------------
// Embedding gather + f32->bf16, producing time-major x0[t][b][d]
// ---------------------------------------------------------------------------
__global__ __launch_bounds__(256) void embed_gather_kernel(
    const int* __restrict__ tokens,      // [B][T]
    const float* __restrict__ emb,       // [V][D]
    unsigned short* __restrict__ x0)     // [T][B][D] bf16
{
  int idx = blockIdx.x * 256 + threadIdx.x;       // over T*B*(D/4)
  int d4  = idx & 63;                             // 64 float4 chunks per row
  int row = idx >> 6;                             // row = t*B + b
  if (row >= T_SEQ * B_BATCH) return;
  int t = row >> 7;                               // /128
  int b = row & 127;
  int tok = tokens[b * T_SEQ + t];
  const float4* src = (const float4*)(emb + (size_t)tok * D_IN);
  float4 v = src[d4];
  ushort4 o;
  o.x = f2bf(v.x); o.y = f2bf(v.y); o.z = f2bf(v.z); o.w = f2bf(v.w);
  *(ushort4*)(&x0[(size_t)row * D_IN + d4 * 4]) = o;
}

// ---------------------------------------------------------------------------
// Weight transpose + convert: WT[n][k] = bf16(W[k][n]), W is [K=256][N=768]
// (B-operand wants contiguous-K per output column for ds_load_b128 fragments)
// ---------------------------------------------------------------------------
__global__ __launch_bounds__(256) void convert_wt_kernel(
    const float* __restrict__ W, unsigned short* __restrict__ WT)
{
  int idx = blockIdx.x * 256 + threadIdx.x;       // over 768*256
  if (idx >= NG3 * U_HID) return;
  int n = idx >> 8;          // /256
  int k = idx & 255;
  WT[idx] = f2bf(W[(size_t)k * NG3 + n]);
}

// ---------------------------------------------------------------------------
// Persistent GRU scan for one layer.
// grid = 16 WGs (one per 16-column block of j in [0,256)), 256 thr = 8 waves.
// Per step: four f32 accumulators via v_wmma_f32_16x16x32_bf16:
//   accZ  = h@Whz + x@Wxz      accR  = h@Whr + x@Wxr
//   accRH = h@Whh              accXH = x@Wxh        (reset gate applies to RH only)
// then GRU nonlinearity, h double-buffered in global bf16, grid barrier.
// x_{t+1} is prefetched (global_prefetch_b8) before the barrier spin so the
// next step's A-fragment loads hit the WGP cache.
// ---------------------------------------------------------------------------
__global__ __launch_bounds__(256) void gru_scan_kernel(
    const unsigned short* __restrict__ xin,   // [T][B][256] bf16 (layer input seq)
    const unsigned short* __restrict__ WhT,   // [768][256] bf16 transposed
    const unsigned short* __restrict__ WxT,   // [768][256] bf16 transposed
    const float* __restrict__ bx,
    const float* __restrict__ bh,
    unsigned short* __restrict__ hsout,       // [T][B][256] bf16 (layer output seq)
    unsigned short* __restrict__ hB,          // [2][B][256] bf16 (h double buffer)
    float* __restrict__ hF,                   // [B][256] f32 running h
    unsigned int* __restrict__ ctr)           // [T] barrier counters (pre-zeroed)
{
  __shared__ __align__(16) unsigned short WhS[48 * 256];  // 24 KB: rows g*16+n
  __shared__ __align__(16) unsigned short WxS[48 * 256];  // 24 KB
  __shared__ float bxS[48], bhS[48];

  const int j0 = blockIdx.x * 16;

  // Stage this WG's 48 weight columns (z,r,h gates) into LDS, transposed [n][k].
  for (int i = threadIdx.x; i < 48 * 16; i += 256) {
    int rl = i >> 4, c = i & 15;            // rl: local row 0..47, c: 16-elem chunk
    int g = rl >> 4, n = rl & 15;
    int grow = g * U_HID + j0 + n;          // row in WT
    *(u32x4*)(&WhS[rl * 256 + c * 16]) = *(const u32x4*)(&WhT[(size_t)grow * 256 + c * 16]);
    *(u32x4*)(&WxS[rl * 256 + c * 16]) = *(const u32x4*)(&WxT[(size_t)grow * 256 + c * 16]);
  }
  if (threadIdx.x < 48) {
    int g = threadIdx.x >> 4, n = threadIdx.x & 15;
    bxS[threadIdx.x] = bx[g * U_HID + j0 + n];
    bhS[threadIdx.x] = bh[g * U_HID + j0 + n];
  }
  __syncthreads();

  const int wave   = threadIdx.x >> 5;       // 0..7  -> M tile (16 rows of b)
  const int lane   = threadIdx.x & 31;
  const int laneLo = lane & 15;
  const int hi     = lane >> 4;              // 0/1: which K half of fragment
  const int rowA   = wave * 16 + laneLo;     // A-fragment row (batch index b)

  // Prefetch first timestep's x rows (cheap; overlaps step 0's recurrent part)
  {
    const unsigned short* pf0 = xin + (size_t)rowA * 256 + hi * 128;
    __builtin_prefetch(pf0, 0, 3);
    __builtin_prefetch(pf0 + 64, 0, 3);
  }

  for (int t = 0; t < T_SEQ; ++t) {
    const unsigned short* hcur = hB + (size_t)(t & 1) * (B_BATCH * U_HID);
    const unsigned short* xt   = xin + (size_t)t * (B_BATCH * U_HID);

    f32x8 accZ = {}, accR = {}, accXH = {}, accRH = {};

    // ---- recurrent part: h_{t-1} @ Wh[:, {z,r,h} cols] ----
    #pragma unroll
    for (int kb = 0; kb < 8; ++kb) {
      FragBF a;
      const unsigned short* ap = hcur + rowA * 256 + kb * 32 + hi * 16;
      a.q[0] = *(const u32x4*)(ap);
      a.q[1] = *(const u32x4*)(ap + 8);
      FragBF b0, b1, b2;
      const unsigned short* bp = &WhS[laneLo * 256 + kb * 32 + hi * 16];
      b0.q[0] = *(const u32x4*)(bp);            b0.q[1] = *(const u32x4*)(bp + 8);
      b1.q[0] = *(const u32x4*)(bp + 16 * 256); b1.q[1] = *(const u32x4*)(bp + 16 * 256 + 8);
      b2.q[0] = *(const u32x4*)(bp + 32 * 256); b2.q[1] = *(const u32x4*)(bp + 32 * 256 + 8);
      accZ  = __builtin_amdgcn_wmma_f32_16x16x32_bf16(false, a.v, false, b0.v, (short)0, accZ,  false, false);
      accR  = __builtin_amdgcn_wmma_f32_16x16x32_bf16(false, a.v, false, b1.v, (short)0, accR,  false, false);
      accRH = __builtin_amdgcn_wmma_f32_16x16x32_bf16(false, a.v, false, b2.v, (short)0, accRH, false, false);
    }
    // ---- input part: x_t @ Wx[:, {z,r,h} cols] ----
    #pragma unroll
    for (int kb = 0; kb < 8; ++kb) {
      FragBF a;
      const unsigned short* ap = xt + rowA * 256 + kb * 32 + hi * 16;
      a.q[0] = *(const u32x4*)(ap);
      a.q[1] = *(const u32x4*)(ap + 8);
      FragBF b0, b1, b2;
      const unsigned short* bp = &WxS[laneLo * 256 + kb * 32 + hi * 16];
      b0.q[0] = *(const u32x4*)(bp);            b0.q[1] = *(const u32x4*)(bp + 8);
      b1.q[0] = *(const u32x4*)(bp + 16 * 256); b1.q[1] = *(const u32x4*)(bp + 16 * 256 + 8);
      b2.q[0] = *(const u32x4*)(bp + 32 * 256); b2.q[1] = *(const u32x4*)(bp + 32 * 256 + 8);
      accZ  = __builtin_amdgcn_wmma_f32_16x16x32_bf16(false, a.v, false, b0.v, (short)0, accZ,  false, false);
      accR  = __builtin_amdgcn_wmma_f32_16x16x32_bf16(false, a.v, false, b1.v, (short)0, accR,  false, false);
      accXH = __builtin_amdgcn_wmma_f32_16x16x32_bf16(false, a.v, false, b2.v, (short)0, accXH, false, false);
    }

    // ---- prefetch next step's x rows so they land in WGP$ during the barrier
    if (t + 1 < T_SEQ) {
      const unsigned short* pf =
          xin + (size_t)(t + 1) * (B_BATCH * U_HID) + (size_t)rowA * 256 + hi * 128;
      __builtin_prefetch(pf, 0, 3);        // bytes [0,128) of this lane's half-row
      __builtin_prefetch(pf + 64, 0, 3);   // bytes [128,256)
    }

    // ---- GRU elementwise update; C/D layout: vgpr r, lanes<16 -> M=r, lanes>=16 -> M=r+8
    const int j     = j0 + laneLo;
    const int bbase = wave * 16 + hi * 8;
    unsigned short* hnext = hB + (size_t)((t + 1) & 1) * (B_BATCH * U_HID);
    unsigned short* hsrow = hsout + (size_t)t * (B_BATCH * U_HID);
    #pragma unroll
    for (int r = 0; r < 8; ++r) {
      int b = bbase + r;
      float z  = sigmoidf_(accZ[r] + bxS[laneLo]      + bhS[laneLo]);
      float rg = sigmoidf_(accR[r] + bxS[16 + laneLo] + bhS[16 + laneLo]);
      float hh = tanhf_(accXH[r] + bxS[32 + laneLo] + rg * (accRH[r] + bhS[32 + laneLo]));
      float hp = hF[b * U_HID + j];
      float hn = z * hp + (1.0f - z) * hh;
      hF[b * U_HID + j] = hn;
      unsigned short hb16 = f2bf(hn);
      hnext[b * U_HID + j] = hb16;
      hsrow[b * U_HID + j] = hb16;
    }

    // ---- device-wide barrier for this timestep ----
    __syncthreads();
    if (threadIdx.x == 0) {
      __threadfence();
      __hip_atomic_fetch_add(&ctr[t], 1u, __ATOMIC_RELEASE, __HIP_MEMORY_SCOPE_AGENT);
      while (__hip_atomic_load(&ctr[t], __ATOMIC_ACQUIRE, __HIP_MEMORY_SCOPE_AGENT)
             < (unsigned)SCAN_BLOCKS)
        __builtin_amdgcn_s_sleep(1);
      __threadfence();
    }
    __syncthreads();
  }
}

// ---------------------------------------------------------------------------
// Dense head: out[b] = sigmoid(relu(h[b]@Wd1 + bd1) @ Wd2 + bd2)
// ---------------------------------------------------------------------------
__global__ __launch_bounds__(256) void dense_head_kernel(
    const float* __restrict__ h,     // [B][256] (final layer-3 hidden state)
    const float* __restrict__ Wd1,   // [256][256]
    const float* __restrict__ bd1,   // [256]
    const float* __restrict__ Wd2,   // [256][1]
    const float* __restrict__ bd2,   // [1]
    float* __restrict__ out)         // [B]
{
  __shared__ float hrow[256];
  __shared__ float red[256];
  const int b = blockIdx.x;
  const int u = threadIdx.x;
  hrow[u] = h[b * U_HID + u];
  __syncthreads();
  float acc = bd1[u];
  for (int k = 0; k < U_HID; ++k) acc += hrow[k] * Wd1[k * U_HID + u];
  red[u] = fmaxf(acc, 0.0f) * Wd2[u];
  __syncthreads();
  for (int s = 128; s > 0; s >>= 1) {
    if (u < s) red[u] += red[u + s];
    __syncthreads();
  }
  if (u == 0) out[b] = 1.0f / (1.0f + __expf(-(red[0] + bd2[0])));
}

// ---------------------------------------------------------------------------
// Host launcher
// ---------------------------------------------------------------------------
extern "C" void kernel_launch(void* const* d_in, const int* in_sizes, int n_in,
                              void* d_out, int out_size, void* d_ws, size_t ws_size,
                              hipStream_t stream) {
  (void)in_sizes; (void)n_in; (void)out_size; (void)ws_size;

  const int*   tokens = (const int*)d_in[0];
  const float* emb    = (const float*)d_in[1];
  const float* Wx[4], *Wh[4], *bx[4], *bh[4];
  for (int l = 0; l < 4; ++l) {
    Wx[l] = (const float*)d_in[2 + 4 * l];
    Wh[l] = (const float*)d_in[3 + 4 * l];
    bx[l] = (const float*)d_in[4 + 4 * l];
    bh[l] = (const float*)d_in[5 + 4 * l];
  }
  const float* Wd1 = (const float*)d_in[18];
  const float* bd1 = (const float*)d_in[19];
  const float* Wd2 = (const float*)d_in[20];
  const float* bd2 = (const float*)d_in[21];
  float* out = (float*)d_out;

  // --- workspace layout (256B-aligned slices) ---
  char* p = (char*)d_ws;
  auto alloc = [&](size_t bytes) -> void* {
    void* r = (void*)p;
    p += (bytes + 255) & ~(size_t)255;
    return r;
  };
  const size_t seqBytes = (size_t)T_SEQ * B_BATCH * U_HID * sizeof(unsigned short); // 32 MB
  unsigned short* seqA = (unsigned short*)alloc(seqBytes);
  unsigned short* seqB = (unsigned short*)alloc(seqBytes);
  unsigned short* WxT[4], *WhT[4];
  for (int l = 0; l < 4; ++l) {
    WxT[l] = (unsigned short*)alloc((size_t)NG3 * U_HID * sizeof(unsigned short));
    WhT[l] = (unsigned short*)alloc((size_t)NG3 * U_HID * sizeof(unsigned short));
  }
  unsigned short* hB = (unsigned short*)alloc((size_t)2 * B_BATCH * U_HID * sizeof(unsigned short));
  float*          hF = (float*)alloc((size_t)B_BATCH * U_HID * sizeof(float));
  unsigned int*  ctr = (unsigned int*)alloc((size_t)4 * T_SEQ * sizeof(unsigned int));

  // barrier counters must be zero each call (deterministic, no cross-call state)
  hipMemsetAsync(ctr, 0, (size_t)4 * T_SEQ * sizeof(unsigned int), stream);

  // weights -> transposed bf16
  const int wtElems = NG3 * U_HID;
  const int wtBlocks = (wtElems + 255) / 256;
  for (int l = 0; l < 4; ++l) {
    convert_wt_kernel<<<wtBlocks, 256, 0, stream>>>(Wx[l], WxT[l]);
    convert_wt_kernel<<<wtBlocks, 256, 0, stream>>>(Wh[l], WhT[l]);
  }

  // embedding gather -> time-major bf16 sequence
  {
    int total = T_SEQ * B_BATCH * (D_IN / 4);
    embed_gather_kernel<<<(total + 255) / 256, 256, 0, stream>>>(tokens, emb, seqA);
  }

  // 4 stacked GRU layers, ping-ponging the sequence buffers
  const unsigned short* inSeq = seqA;
  unsigned short* outSeq = seqB;
  for (int l = 0; l < 4; ++l) {
    hipMemsetAsync(hB, 0, (size_t)2 * B_BATCH * U_HID * sizeof(unsigned short), stream);
    hipMemsetAsync(hF, 0, (size_t)B_BATCH * U_HID * sizeof(float), stream);
    gru_scan_kernel<<<SCAN_BLOCKS, 256, 0, stream>>>(
        inSeq, WhT[l], WxT[l], bx[l], bh[l], outSeq, hB, hF, ctr + (size_t)l * T_SEQ);
    const unsigned short* tmp = inSeq;
    inSeq = outSeq;
    outSeq = (unsigned short*)tmp;
  }

  // final dense head on the layer-3 last hidden state (lives in hF)
  dense_head_kernel<<<B_BATCH, 256, 0, stream>>>(hF, Wd1, bd1, Wd2, bd2, out);
}
